// FAVORPlusAttention_28698971472315
// MI455X (gfx1250) — matrix operand order repaired and checked by reading, not compiled, not run on hardware
//
#include <hip/hip_runtime.h>

// ---------------- problem constants ----------------
#define Bsz 4
#define Ssz 4096
#define Esz 1024
#define Hsz 16
#define Dsz 64
#define Msz 256
#define BHsz (Bsz*Hsz)

__device__ __constant__ float DNORM = 0.35355339059327373f; // 64^-0.25
__device__ __constant__ float RATIO = 0.0625f;              // 256^-0.5
__device__ __constant__ float EPSF  = 1e-4f;

// ---------------- WMMA types / helpers ----------------
typedef __attribute__((ext_vector_type(16))) __bf16 v16bf;
typedef __attribute__((ext_vector_type(8)))  float  v8f;

__device__ __forceinline__ v8f zero8() {
    v8f z;
#pragma unroll
    for (int i = 0; i < 8; ++i) z[i] = 0.0f;
    return z;
}

__device__ __forceinline__ v8f wmma_bf16(v16bf a, v16bf b, v8f c) {
    return __builtin_amdgcn_wmma_f32_16x16x32_bf16(
        /*neg_a=*/false, a, /*neg_b=*/false, b,
        /*c_mod=*/(short)0, c, /*reuse_a=*/false, /*reuse_b=*/false);
}

// A-matrix (16x32 bf16): row = lane&15, elem e: K = ((e&8)<<1)+(e&7)+8*(lane>=16)
__device__ __forceinline__ int a_k(int lane, int e) {
    int hi = (lane >> 4) & 1;
    return ((e & 8) << 1) + (e & 7) + (hi << 3);
}
// B-matrix (32x16 bf16): col = lane&15, elem e: K = ((e&8)<<1)+(e&7)+8*(lane>=16)
__device__ __forceinline__ int b_k(int lane, int e) {
    int hi = (lane >> 4) & 1;
    return ((e & 8) << 1) + (e & 7) + (hi << 3);
}
// C/D f32 16x16: elem r: row = r + 8*(lane>=16), col = lane&15
__device__ __forceinline__ int c_row(int lane, int r) {
    return r + (((lane >> 4) & 1) << 3);
}

__device__ __forceinline__ __bf16 f2bf(float f) { return (__bf16)f; }
__device__ __forceinline__ unsigned short bf2us(__bf16 b) {
    return __builtin_bit_cast(unsigned short, b);
}
__device__ __forceinline__ __bf16 us2bf(unsigned short u) {
    return __builtin_bit_cast(__bf16, u);
}

// ---- gfx1250 async global->LDS copy (ASYNCcnt path) ----
// 16 bytes per lane: LDS[vdst] = MEM[vaddr]; generic LDS ptr low 32 bits = LDS offset.
__device__ __forceinline__ void async_g2l_b128(void* lds_ptr, const void* gptr) {
    unsigned loff = (unsigned)(size_t)lds_ptr;
    asm volatile("global_load_async_to_lds_b128 %0, %1, off"
                 :: "v"(loff), "v"(gptr)
                 : "memory");
}
template <int N>
__device__ __forceinline__ void async_wait() {
#if __has_builtin(__builtin_amdgcn_s_wait_asynccnt)
    __builtin_amdgcn_s_wait_asynccnt((short)N);
#else
    asm volatile("s_wait_asynccnt %0" :: "i"(N) : "memory");
#endif
}

// monotone float<->uint map for atomicMax over signed floats
__device__ __forceinline__ unsigned f2ord(float f) {
    unsigned u = __float_as_uint(f);
    return (u & 0x80000000u) ? ~u : (u | 0x80000000u);
}
__device__ __forceinline__ float ord2f(unsigned u) {
    unsigned b = (u & 0x80000000u) ? (u & 0x7fffffffu) : ~u;
    return __uint_as_float(b);
}

// ---------------- kernel 0: init accumulators ----------------
__global__ void __launch_bounds__(256) k_init(float* kv, float* ksum, unsigned* stab) {
    int i = blockIdx.x * 256 + threadIdx.x;
    if (i < BHsz * Msz * Dsz) kv[i] = 0.0f;
    if (i < BHsz * Msz) ksum[i] = 0.0f;
    if (i == 0) *stab = 0u;
}

// ---------------- kernel 1: global stabilizer for K ----------------
// grid: BH * (S/64); 256 threads (8 waves). Wave w: s-tile w&3, m-tiles (w>>2)*8+j.
__global__ void __launch_bounds__(256) k_kstab(const float* __restrict__ kin,
                                               const float* __restrict__ proj,
                                               unsigned* __restrict__ stab_ord) {
    const int nsc = Ssz / 64;
    int bh = blockIdx.x / nsc;
    int s0 = (blockIdx.x % nsc) * 64;
    int b = bh / Hsz, h = bh % Hsz;
    int tid = threadIdx.x, wave = tid >> 5, lane = tid & 31;

    const float* kbase = kin + (size_t)b * Ssz * Esz + (size_t)h * Dsz;
    int srow = s0 + (wave & 3) * 16 + (lane & 15);

    v16bf afrag[2];
#pragma unroll
    for (int ks = 0; ks < 2; ++ks) {
        const float* rp = kbase + (size_t)srow * Esz + ks * 32;
#pragma unroll
        for (int e = 0; e < 16; ++e) afrag[ks][e] = f2bf(rp[a_k(lane, e)] * DNORM);
    }

    float lmax = -3.0e38f;
    for (int j = 0; j < 8; ++j) {
        int m0 = ((wave >> 2) * 8 + j) * 16;
        v8f c = zero8();
#pragma unroll
        for (int ks = 0; ks < 2; ++ks) {
            v16bf bfr;
            const float* pp = proj + (size_t)(m0 + (lane & 15)) * Dsz + ks * 32;
#pragma unroll
            for (int e = 0; e < 16; ++e) bfr[e] = f2bf(pp[b_k(lane, e)]);
            c = wmma_bf16(afrag[ks], bfr, c);
        }
#pragma unroll
        for (int r = 0; r < 8; ++r) lmax = fmaxf(lmax, c[r]);
    }

    __shared__ float red[256];
    red[tid] = lmax;
    __syncthreads();
    for (int off = 128; off; off >>= 1) {
        if (tid < off) red[tid] = fmaxf(red[tid], red[tid + off]);
        __syncthreads();
    }
    if (tid == 0) atomicMax(stab_ord, f2ord(red[0]));
}

// ---------------- kernel 2: kv = k'^T v, k_sum = sum_s k' ----------------
// grid: BH * (S/CHUNK); 256 threads (8 waves)
#define CHUNK 512
__global__ void __launch_bounds__(256) k_kv(const float* __restrict__ kin,
                                            const float* __restrict__ vin,
                                            const float* __restrict__ proj,
                                            const unsigned* __restrict__ stab_ord,
                                            float* __restrict__ kv,
                                            float* __restrict__ ksum) {
    const int nch = Ssz / CHUNK;
    int bh = blockIdx.x / nch;
    int sbeg = (blockIdx.x % nch) * CHUNK;
    int b = bh / Hsz, h = bh % Hsz;
    int tid = threadIdx.x, wave = tid >> 5, lane = tid & 31;
    float stab = ord2f(*stab_ord);

    __shared__ unsigned short kbuf[32][Msz + 8]; // k' bf16, [s_local][m]
    __shared__ unsigned short vbuf[32][Dsz + 8]; // v  bf16, [s_local][d]
    __shared__ float dpart[32][8];
    __shared__ float diag[32];

    v8f acc[8]; // wave owns m-tiles (2w,2w+1) x d-tiles 0..3
#pragma unroll
    for (int i = 0; i < 8; ++i) acc[i] = zero8();
    float ksum_acc = 0.0f; // thread tid owns feature m = tid

    const float* kpB = kin + (size_t)b * Ssz * Esz + (size_t)h * Dsz;
    const float* vpB = vin + (size_t)b * Ssz * Esz + (size_t)h * Dsz;

    for (int s0 = sbeg; s0 < sbeg + CHUNK; s0 += 32) {
        // diag partials + stage v into LDS (bf16); prefetch next s-step
        {
            int r = tid >> 3, p = tid & 7;
            const float* rp = kpB + (size_t)(s0 + r) * Esz + p * 8;
            const float* vr = vpB + (size_t)(s0 + r) * Esz + p * 8;
            if (s0 + 32 < sbeg + CHUNK) {
                __builtin_prefetch(rp + 32 * Esz, 0, 0);
                __builtin_prefetch(vr + 32 * Esz, 0, 0);
            }
            float ds = 0.0f;
#pragma unroll
            for (int e = 0; e < 8; ++e) {
                float x = rp[e] * DNORM;
                ds += x * x;
                vbuf[r][p * 8 + e] = bf2us(f2bf(vr[e]));
            }
            dpart[r][p] = 0.5f * ds;
        }
        __syncthreads();
        if (tid < 32) {
            float s = 0.0f;
#pragma unroll
            for (int p = 0; p < 8; ++p) s += dpart[tid][p];
            diag[tid] = s;
        }
        __syncthreads();

        // k' tiles via WMMA: wave w -> s-tile w&1, m-tiles (w>>1)*4 + j
        {
            int stile = wave & 1;
            int srow = s0 + stile * 16 + (lane & 15);
            v16bf afrag[2];
#pragma unroll
            for (int ks = 0; ks < 2; ++ks) {
                const float* rp = kpB + (size_t)srow * Esz + ks * 32;
#pragma unroll
                for (int e = 0; e < 16; ++e) afrag[ks][e] = f2bf(rp[a_k(lane, e)] * DNORM);
            }
            for (int j = 0; j < 4; ++j) {
                int m0 = ((wave >> 1) * 4 + j) * 16;
                v8f c = zero8();
#pragma unroll
                for (int ks = 0; ks < 2; ++ks) {
                    v16bf bfr;
                    const float* pp = proj + (size_t)(m0 + (lane & 15)) * Dsz + ks * 32;
#pragma unroll
                    for (int e = 0; e < 16; ++e) bfr[e] = f2bf(pp[b_k(lane, e)]);
                    c = wmma_bf16(afrag[ks], bfr, c);
                }
#pragma unroll
                for (int r = 0; r < 8; ++r) {
                    int row = stile * 16 + c_row(lane, r);
                    float val = RATIO * (__expf(c[r] - diag[row] - stab) + EPSF);
                    kbuf[row][m0 + (lane & 15)] = bf2us(f2bf(val));
                }
            }
        }
        __syncthreads();

        // k_sum accumulation: thread tid sums column m=tid over 32 rows
#pragma unroll 8
        for (int r = 0; r < 32; ++r) ksum_acc += (float)us2bf(kbuf[r][tid]);

        // kv += k'^T v  (A[m][k=s] from LDS transpose, B[k=s][d] from LDS)
        for (int mt = 0; mt < 2; ++mt) {
            int m0 = (wave * 2 + mt) * 16;
            v16bf afr;
#pragma unroll
            for (int e = 0; e < 16; ++e)
                afr[e] = us2bf(kbuf[a_k(lane, e)][m0 + (lane & 15)]);
            for (int dt = 0; dt < 4; ++dt) {
                int d0 = dt * 16;
                v16bf bfr;
#pragma unroll
                for (int e = 0; e < 16; ++e)
                    bfr[e] = us2bf(vbuf[b_k(lane, e)][d0 + (lane & 15)]);
                acc[mt * 4 + dt] = wmma_bf16(afr, bfr, acc[mt * 4 + dt]);
            }
        }
        __syncthreads();
    }

    atomicAdd(&ksum[bh * Msz + tid], ksum_acc);
    float* kvb = kv + (size_t)bh * Msz * Dsz;
    for (int mt = 0; mt < 2; ++mt)
        for (int dt = 0; dt < 4; ++dt) {
            int m0 = (wave * 2 + mt) * 16, d0 = dt * 16;
            v8f c = acc[mt * 4 + dt];
#pragma unroll
            for (int r = 0; r < 8; ++r)
                atomicAdd(&kvb[(size_t)(m0 + c_row(lane, r)) * Dsz + d0 + (lane & 15)], c[r]);
        }
}

// ---------------- kernel 2b: kv f32 -> bf16 (once) ----------------
__global__ void __launch_bounds__(256) k_kvcvt(const float* __restrict__ kv,
                                               unsigned short* __restrict__ kvbf) {
    int i = blockIdx.x * 256 + threadIdx.x;
    if (i < BHsz * Msz * Dsz) kvbf[i] = bf2us(f2bf(kv[i]));
}

// ---------------- kernel 3: q' (fused) -> context (bf16) ----------------
// grid: BH * (S/32); 256 threads (8 waves). kv tile staged via async global->LDS,
// overlapped with the dd_q WMMA phase.
__global__ void __launch_bounds__(256) k_qctx(const float* __restrict__ qin,
                                              const float* __restrict__ proj,
                                              const unsigned short* __restrict__ kvbf,
                                              const float* __restrict__ ksum,
                                              unsigned short* __restrict__ ctx) {
    const int nst = Ssz / 32;
    int bh = blockIdx.x / nst;
    int s0 = (blockIdx.x % nst) * 32;
    int b = bh / Hsz, h = bh % Hsz;
    int tid = threadIdx.x, wave = tid >> 5, lane = tid & 31;

    __shared__ float dd[32][Msz];           // 32 KB
    __shared__ unsigned short qp[32][Msz];  // 16 KB (q' bf16)
    __shared__ unsigned short kvlds[Msz * Dsz]; // 32 KB (kv bf16, [m][d])
    __shared__ float part[32][8];
    __shared__ float diag[32], rmax[32], denom[32];

    // ---- kick off async copy of kv[bh] into LDS (8 x b128 per thread) ----
    {
        const char* gsrc = (const char*)(kvbf + (size_t)bh * Msz * Dsz);
        char* ldst = (char*)kvlds;
#pragma unroll
        for (int j = 0; j < 8; ++j) {
            int i = (tid * 8 + j) * 16; // byte offset, 32 KB total
            async_g2l_b128(ldst + i, gsrc + i);
        }
    }

    const float* qpB = qin + (size_t)b * Ssz * Esz + (size_t)h * Dsz;

    // diag: 256 threads, row = tid>>3, part = tid&7
    {
        int r = tid >> 3, p = tid & 7;
        const float* rp = qpB + (size_t)(s0 + r) * Esz + p * 8;
        float s = 0.0f;
#pragma unroll
        for (int e = 0; e < 8; ++e) { float x = rp[e] * DNORM; s += x * x; }
        part[r][p] = 0.5f * s;
    }
    __syncthreads();
    if (tid < 32) {
        float s = 0.0f;
#pragma unroll
        for (int p = 0; p < 8; ++p) s += part[tid][p];
        diag[tid] = s;
    }
    __syncthreads();

    // dd tiles via WMMA: wave w -> s-tile w&1, m-tiles (w>>1)*4 + j
    {
        int stile = wave & 1;
        int srow = s0 + stile * 16 + (lane & 15);
        v16bf afrag[2];
#pragma unroll
        for (int ks = 0; ks < 2; ++ks) {
            const float* rp = qpB + (size_t)srow * Esz + ks * 32;
#pragma unroll
            for (int e = 0; e < 16; ++e) afrag[ks][e] = f2bf(rp[a_k(lane, e)] * DNORM);
        }
        for (int j = 0; j < 4; ++j) {
            int m0 = ((wave >> 1) * 4 + j) * 16;
            v8f c = zero8();
#pragma unroll
            for (int ks = 0; ks < 2; ++ks) {
                v16bf bfr;
                const float* pp = proj + (size_t)(m0 + (lane & 15)) * Dsz + ks * 32;
#pragma unroll
                for (int e = 0; e < 16; ++e) bfr[e] = f2bf(pp[b_k(lane, e)]);
                c = wmma_bf16(afrag[ks], bfr, c);
            }
#pragma unroll
            for (int r = 0; r < 8; ++r)
                dd[stile * 16 + c_row(lane, r)][m0 + (lane & 15)] = c[r];
        }
    }
    __syncthreads();

    // per-row max: row = tid&31, p = tid>>5 (8 chunks of 32 cols)
    {
        int r = tid & 31, p = tid >> 5;
        float m = -3.0e38f;
        for (int c = p * 32; c < p * 32 + 32; ++c) m = fmaxf(m, dd[r][c]);
        part[r][p] = m;
    }
    __syncthreads();
    if (tid < 32) {
        float m = -3.0e38f;
#pragma unroll
        for (int p = 0; p < 8; ++p) m = fmaxf(m, part[tid][p]);
        rmax[tid] = m;
    }
    __syncthreads();

    // q' + denominator partials
    {
        int r = tid & 31, p = tid >> 5;
        const float* ksb = ksum + bh * Msz;
        float dsum = 0.0f;
        for (int c = p * 32; c < p * 32 + 32; ++c) {
            float v = RATIO * (__expf(dd[r][c] - diag[r] - rmax[r]) + EPSF);
            qp[r][c] = bf2us(f2bf(v));
            dsum += v * ksb[c];
        }
        part[r][p] = dsum;
    }
    __syncthreads();
    if (tid < 32) {
        float s = 0.0f;
#pragma unroll
        for (int p = 0; p < 8; ++p) s += part[tid][p];
        denom[tid] = s;
    }
    // async kv copy must be complete before the num GEMM consumes kvlds
    async_wait<0>();
    __syncthreads();

    // num = q' * kv ; wave w -> s-tile w>>2, d-tile w&3; K=256 in 8 steps
    {
        int stile = wave >> 2;
        int d0 = (wave & 3) * 16;
        v8f c = zero8();
        for (int ks = 0; ks < 8; ++ks) {
            v16bf afr, bfr;
#pragma unroll
            for (int e = 0; e < 16; ++e)
                afr[e] = us2bf(qp[stile * 16 + (lane & 15)][ks * 32 + a_k(lane, e)]);
#pragma unroll
            for (int e = 0; e < 16; ++e)
                bfr[e] = us2bf(kvlds[(ks * 32 + b_k(lane, e)) * Dsz + d0 + (lane & 15)]);
            c = wmma_bf16(afr, bfr, c);
        }
#pragma unroll
        for (int r = 0; r < 8; ++r) {
            int row = stile * 16 + c_row(lane, r);
            int s = s0 + row;
            int e = h * Dsz + d0 + (lane & 15);
            float val = c[r] / denom[row];
            ctx[(size_t)(b * Ssz + s) * Esz + e] = bf2us(f2bf(val));
        }
    }
}

// ---------------- kernel 4: out = ctx @ W^T + b ----------------
// grid: (N/64, E/64); 256 threads (8 waves). Software-pipelined (double-buffered)
// over K: ctx tile comes in via async global->LDS b128 (ASYNCcnt), next tile's
// copies are issued before computing the current one; wait asynccnt<=1 forces
// only the current buffer to completion while the next stays in flight.
__global__ void __launch_bounds__(256) k_out(const unsigned short* __restrict__ ctx,
                                             const float* __restrict__ W,
                                             const float* __restrict__ bias,
                                             float* __restrict__ out) {
    int n0 = blockIdx.x * 64, e0 = blockIdx.y * 64;
    int tid = threadIdx.x, wave = tid >> 5, lane = tid & 31;

    __shared__ unsigned short abuf[2][64][32]; // [buf][n][k], 64B rows (16B aligned)
    __shared__ unsigned short bbuf[2][32][72]; // [buf][k][eo] (64 + pad)

    int nt = wave >> 1;        // n-tile 0..3
    int et0 = (wave & 1) * 2;  // e-tiles et0, et0+1
    int ar = tid >> 2, ap = tid & 3; // ctx staging: row, 16B chunk
    int eo = tid >> 2, wp = tid & 3; // W staging: output col, 8-elem chunk

    v8f acc[2] = {zero8(), zero8()};

    // prologue: stage k-step 0 into buffer 0
    async_g2l_b128(&abuf[0][ar][ap * 8],
                   ctx + (size_t)(n0 + ar) * Esz + ap * 8);
    {
        const float* src = W + (size_t)(e0 + eo) * Esz + wp * 8;
#pragma unroll
        for (int e = 0; e < 8; ++e) bbuf[0][wp * 8 + e][eo] = bf2us(f2bf(src[e]));
    }

    const int NSTEP = Esz / 32;
    for (int i = 0; i < NSTEP; ++i) {
        int cur = i & 1;
        // stage next k-step into the other buffer (async + manual, both overlap)
        if (i + 1 < NSTEP) {
            int kk = (i + 1) * 32;
            async_g2l_b128(&abuf[cur ^ 1][ar][ap * 8],
                           ctx + (size_t)(n0 + ar) * Esz + kk + ap * 8);
            const float* src = W + (size_t)(e0 + eo) * Esz + kk + wp * 8;
#pragma unroll
            for (int e = 0; e < 8; ++e)
                bbuf[cur ^ 1][wp * 8 + e][eo] = bf2us(f2bf(src[e]));
            async_wait<1>(); // current buffer complete; next may stay in flight
        } else {
            async_wait<0>();
        }
        __syncthreads();

        v16bf afr;
#pragma unroll
        for (int e = 0; e < 16; ++e)
            afr[e] = us2bf(abuf[cur][nt * 16 + (lane & 15)][a_k(lane, e)]);
#pragma unroll
        for (int j = 0; j < 2; ++j) {
            v16bf bfr;
            int ecol = (et0 + j) * 16 + (lane & 15);
#pragma unroll
            for (int e = 0; e < 16; ++e) bfr[e] = us2bf(bbuf[cur][b_k(lane, e)][ecol]);
            acc[j] = wmma_bf16(afr, bfr, acc[j]);
        }
        __syncthreads();
    }

#pragma unroll
    for (int j = 0; j < 2; ++j) {
#pragma unroll
        for (int r = 0; r < 8; ++r) {
            int n = n0 + nt * 16 + c_row(lane, r);
            int e = e0 + (et0 + j) * 16 + (lane & 15);
            out[(size_t)n * Esz + e] = acc[j][r] + bias[e];
        }
    }
}

// ---------------- host launcher ----------------
extern "C" void kernel_launch(void* const* d_in, const int* in_sizes, int n_in,
                              void* d_out, int out_size, void* d_ws, size_t ws_size,
                              hipStream_t stream) {
    (void)in_sizes; (void)n_in; (void)out_size; (void)ws_size;
    const float* q    = (const float*)d_in[0];
    const float* k    = (const float*)d_in[1];
    const float* v    = (const float*)d_in[2];
    const float* proj = (const float*)d_in[3];
    const float* W    = (const float*)d_in[4];
    const float* bias = (const float*)d_in[5];
    float* out = (float*)d_out;

    char* ws = (char*)d_ws;
    size_t off = 0;
    unsigned* stab = (unsigned*)(ws + off); off += 256;
    float* kv = (float*)(ws + off);         off += (size_t)BHsz * Msz * Dsz * sizeof(float); // 4 MB
    float* ksum = (float*)(ws + off);       off += (size_t)BHsz * Msz * sizeof(float);       // 64 KB
    unsigned short* kvbf = (unsigned short*)(ws + off);
    off += (size_t)BHsz * Msz * Dsz * sizeof(unsigned short);                                // 2 MB
    unsigned short* ctx = (unsigned short*)(ws + off);                                       // 32 MB bf16

    const int nkv = BHsz * Msz * Dsz;
    k_init<<<(nkv + 255) / 256, 256, 0, stream>>>(kv, ksum, stab);
    k_kstab<<<BHsz * (Ssz / 64), 256, 0, stream>>>(k, proj, stab);
    k_kv<<<BHsz * (Ssz / CHUNK), 256, 0, stream>>>(k, v, proj, stab, kv, ksum);
    k_kvcvt<<<(nkv + 255) / 256, 256, 0, stream>>>(kv, kvbf);
    k_qctx<<<BHsz * (Ssz / 32), 256, 0, stream>>>(q, proj, kvbf, ksum, ctx);
    dim3 g4(Bsz * Ssz / 64, Esz / 64);
    k_out<<<g4, 256, 0, stream>>>(ctx, W, bias, out);
}
